// FlatConv3x44x3NNUE_47519518163394
// MI455X (gfx1250) — matrix operand order repaired and checked by reading, not compiled, not run on hardware
//
#include <hip/hip_runtime.h>
#include <hip/hip_bf16.h>

// ---------------------------------------------------------------------------
// MI455X (gfx1250, wave32) implementation.
// All conv layers are im2col GEMMs on v_wmma_f32_16x16x32_f16.
// ---------------------------------------------------------------------------

typedef __attribute__((ext_vector_type(16))) _Float16 v16h;
typedef __attribute__((ext_vector_type(8)))  _Float16 v8h;
typedef __attribute__((ext_vector_type(8)))  float    v8f;

#define NB        4096
#define NSTACK    40
#define BM        16          // samples per workgroup
#define QMAXF     0.9921875f  // 127/128

// workspace layout (bytes)
constexpr size_t OFF_WT0  = 0;                                   // 40*128*32  f16
constexpr size_t OFF_WT1  = OFF_WT0 + 40ull * 128 * 32  * 2;     // 40*128*512 f16
constexpr size_t OFF_WT2  = OFF_WT1 + 40ull * 128 * 512 * 2;     // 40*128*256 f16
constexpr size_t OFF_WT3  = OFF_WT2 + 40ull * 128 * 256 * 2;     // 40*32*128  f16
constexpr size_t OFF_BN   = OFF_WT3 + 40ull * 32  * 128 * 2;     // 40*4*2*128 f32
constexpr size_t OFF_FEAT = OFF_BN  + 40ull * 4 * 2 * 128 * 4;   // 4096*64    f32

struct StackParams {            // one conv-stack group (20 stacks)
  const float* b[4];
  const float* be[4];
  const float* g[4];
  const float* m[4];
  const float* v[4];
  const float* w[4];
};

// mish(v) = v * tanh(softplus(v)).  With e = exp(v):
//   tanh(ln(1+e)) = ((1+e)^2 - 1) / ((1+e)^2 + 1) = e(e+2) / (e(e+2)+2)
// One v_exp_f32 + two FMAs + one v_rcp_f32; exp arg clamped at 15 so the
// rational form never overflows (t = 1 - ~2e-13 there). Branch-free.
__device__ __forceinline__ float mishf(float v) {
  float e = __expf(fminf(v, 15.f));
  float n = e * (e + 2.f);
  float t = n * __builtin_amdgcn_rcpf(n + 2.f);
  return v * t;
}
__device__ __forceinline__ float fq(float v) {   // fake_quant fwd = round(v*128)/128
  return rintf(v * 128.0f) * 0.0078125f;
}
// Assemble the 16-half A fragment from two contiguous 8-half runs
// (per the CDNA5 16-bit A 16x32 lane layout: K = 16*(j/4) + 8*half + 2*(j%4)).
__device__ __forceinline__ v16h combine8(const _Float16* p0, const _Float16* p1) {
  v8h lo = *(const v8h*)p0;
  v8h hi = *(const v8h*)p1;
  return __builtin_shufflevector(lo, hi, 0,1,2,3,4,5,6,7,8,9,10,11,12,13,14,15);
}
__device__ __forceinline__ v8f wmma32(v16h a, v16h b, v8f c) {
  return __builtin_amdgcn_wmma_f32_16x16x32_f16(false, a, false, b, (short)0, c,
                                                false, false);
}

// ---------------------------------------------------------------------------
// prep: weights -> f16 [co][k] with k = (window_pos)*128 + cin  (contiguous
// per B-fragment lane), and fold bias+BN into y = A*conv + C per channel.
// grid: (4 layers, 40 stacks) x 256 threads
// ---------------------------------------------------------------------------
__global__ __launch_bounds__(256) void prep_kernel(
    StackParams P34, StackParams P43,
    _Float16* __restrict__ wt0, _Float16* __restrict__ wt1,
    _Float16* __restrict__ wt2, _Float16* __restrict__ wt3,
    float* __restrict__ bn)
{
  const int l     = blockIdx.x;
  const int stack = blockIdx.y;
  const bool ori34 = stack < 20;
  const StackParams P = ori34 ? P34 : P43;
  const int gi  = ori34 ? stack : stack - 20;
  const int tid = threadIdx.x;

  if (l == 0) {                       // [128][2][2][2] -> [128][32] (K pad 8->32)
    const float* src = P.w[0] + (size_t)gi * 128 * 8;
    _Float16* dst = wt0 + (size_t)stack * 128 * 32;
    for (int e = tid; e < 128 * 32; e += 256) {
      int co = e >> 5, k = e & 31;
      dst[e] = (_Float16)(k < 8 ? src[co * 8 + k] : 0.f);
    }
  } else if (l == 1) {                // [128][128][2][2] -> [128][q*128+ci]
    const float* src = P.w[1] + (size_t)gi * 128 * 512;
    _Float16* dst = wt1 + (size_t)stack * 128 * 512;
    for (int e = tid; e < 128 * 512; e += 256) {
      int co = e >> 9, k = e & 511, q = k >> 7, ci = k & 127;
      dst[e] = (_Float16)src[co * 512 + ci * 4 + q];
    }
  } else if (l == 2) {                // [128][128][2] -> [128][q*128+ci]
    const float* src = P.w[2] + (size_t)gi * 128 * 256;
    _Float16* dst = wt2 + (size_t)stack * 128 * 256;
    for (int e = tid; e < 128 * 256; e += 256) {
      int co = e >> 8, k = e & 255, q = k >> 7, ci = k & 127;
      dst[e] = (_Float16)src[co * 256 + ci * 2 + q];
    }
  } else {                            // [32][128] 1x1
    const float* src = P.w[3] + (size_t)gi * 32 * 128;
    _Float16* dst = wt3 + (size_t)stack * 32 * 128;
    for (int e = tid; e < 32 * 128; e += 256) dst[e] = (_Float16)src[e];
  }

  const int co_n = (l < 3) ? 128 : 32;
  float* bnL = bn + ((size_t)stack * 4 + l) * 2 * 128;
  for (int ch = tid; ch < 128; ch += 256) {
    float A = 0.f, C = 0.f;
    if (ch < co_n) {
      int o = gi * co_n + ch;
      float gg = P.g[l][o], vv = P.v[l][o];
      float bb = P.b[l][o], mm = P.m[l][o], be = P.be[l][o];
      A = gg * rsqrtf(vv + 1e-5f);
      C = A * (bb - mm) + be;
    }
    bnL[ch] = A;
    bnL[128 + ch] = C;
  }
}

// ---------------------------------------------------------------------------
// zero feature accumulator + policy output
// ---------------------------------------------------------------------------
__global__ __launch_bounds__(256) void zero_kernel(float* __restrict__ feat,
                                                   float* __restrict__ pol,
                                                   int nf, int np)
{
  int i = blockIdx.x * blockDim.x + threadIdx.x;
  if (i < nf) feat[i] = 0.f;
  if (i < np) pol[i]  = 0.f;
}

// ---------------------------------------------------------------------------
// One workgroup = (stack, 16-sample chunk). 256 threads = 8 wave32.
// Full 4-layer pipeline in LDS with WMMA GEMMs; result clipped, quantized to
// 1/128 and atomically accumulated into feat[B][64].
// L0 im2col rows are POS-MAJOR (m = pos*16 + s) so epilogue index math is
// pure bit ops (pos = mt, s = 8*hh + r) -- no integer division.
// grid: (256 chunks, 40 stacks)
// ---------------------------------------------------------------------------
__global__ __launch_bounds__(256) void stack_kernel(
    const float* __restrict__ x,
    const _Float16* __restrict__ wt0, const _Float16* __restrict__ wt1,
    const _Float16* __restrict__ wt2, const _Float16* __restrict__ wt3,
    const float* __restrict__ bn, float* __restrict__ feat)
{
  const int stack = blockIdx.y;
  const int s0    = blockIdx.x * BM;
  const bool ori34 = stack < 20;
  const int tidx  = ori34 ? stack : stack - 20;
  const int ty    = ori34 ? (tidx >> 2) : (tidx / 5);
  const int tx    = ori34 ? (tidx & 3)  : (tidx % 5);
  const int W0out = ori34 ? 3 : 2;      // width of the 6-position L0 grid

  const int tid  = threadIdx.x;
  const int wave = tid >> 5;
  const int lane = tid & 31;
  const int ln   = lane & 15;
  const int hh   = lane >> 4;

  __shared__ __align__(16) _Float16 sA0[96 * 32];          //  6 KB  L0 im2col
  __shared__ __align__(16) _Float16 sAct1[6 * BM * 128];   // 24 KB  L0 out [pos][s][ch]
  __shared__ __align__(16) _Float16 sAct2[2 * BM * 128];   //  8 KB  L1 out [p][s][ch]
  __shared__ __align__(16) _Float16 sAct3[BM * 128];       //  4 KB  L2 out [s][ch]

  // ---- stage 0: build L0 im2col  (row m = pos*16 + s, K=8 real, pad to 32)
  for (int idx = tid; idx < 96 * 32; idx += 256) {
    int m = idx >> 5, k = idx & 31;
    float v = 0.f;
    if (k < 8) {
      int pos = m >> 4, s = m & 15;
      int oy = pos / W0out, ox = pos % W0out;
      int ci = k >> 2, dy = (k >> 1) & 1, dx = k & 1;
      int gs = s0 + s;
      v = x[(((size_t)gs * 2 + ci) * 7 + (ty + oy + dy)) * 7 + (tx + ox + dx)];
    }
    sA0[idx] = (_Float16)v;
  }
  __syncthreads();

  // ---- stage 1: L0 GEMM  M=96 N=128 K=32, 48 tiles (mt == pos)
  {
    const float* bnL = bn + ((size_t)stack * 4 + 0) * 2 * 128;
    for (int t = wave; t < 48; t += 8) {
      int mt = t >> 3, nt = t & 7;
      const _Float16* pa = sA0 + (mt * 16 + ln) * 32;
      v16h a = combine8(pa + 8 * hh, pa + 16 + 8 * hh);
      int ch = nt * 16 + ln;
      v16h b = combine8(wt0 + ((size_t)stack * 128 + ch) * 32 + 8 * hh, wt0 + ((size_t)stack * 128 + ch) * 32 + 16 + 8 * hh);
      v8f c = {};
      c = wmma32(a, b, c);
      float A = bnL[ch], C = bnL[128 + ch];
      #pragma unroll
      for (int r = 0; r < 8; ++r) {
        int mm = mt * 16 + 8 * hh + r;        // = pos*BM + s directly
        sAct1[mm * 128 + ch] = (_Float16)mishf(A * c[r] + C);
      }
    }
  }
  __syncthreads();

  // ---- stage 2: L1 GEMM  M=32 (s*2+p) N=128 K=512 (q*128+c), 16 tiles x16 k
  {
    const float* bnL = bn + ((size_t)stack * 4 + 1) * 2 * 128;
    for (int t = wave; t < 16; t += 8) {
      int mt = t >> 3, nt = t & 7;
      int mA = mt * 16 + ln, s = mA >> 1, p = mA & 1;
      int ch = nt * 16 + ln;
      const _Float16* pb = wt1 + ((size_t)stack * 128 + ch) * 512;
      v8f c = {};
      #pragma unroll
      for (int ks = 0; ks < 16; ++ks) {
        int kbase = ks * 32;
        int q = kbase >> 7;   // window element dy*2+dx
        int pos = ori34 ? ((q >> 1) * 3 + p + (q & 1))
                        : ((p + (q >> 1)) * 2 + (q & 1));
        int cb = (kbase & 127) + 8 * hh;
        const _Float16* pa = sAct1 + (pos * BM + s) * 128;
        v16h a = combine8(pa + cb, pa + cb + 16);
        v16h b = combine8(pb + kbase + 8 * hh, pb + kbase + 16 + 8 * hh);
        c = wmma32(a, b, c);
      }
      float A = bnL[ch], C = bnL[128 + ch];
      #pragma unroll
      for (int r = 0; r < 8; ++r) {
        int mm = mt * 16 + 8 * hh + r;
        sAct2[((mm & 1) * BM + (mm >> 1)) * 128 + ch] =
            (_Float16)mishf(A * c[r] + C);
      }
    }
  }
  __syncthreads();

  // ---- stage 3: L2 GEMM  M=16 N=128 K=256 (q=L1 position), 8 tiles x8 k
  {
    const float* bnL = bn + ((size_t)stack * 4 + 2) * 2 * 128;
    int nt = wave;                       // exactly 8 tiles
    int ch = nt * 16 + ln;
    const _Float16* pb = wt2 + ((size_t)stack * 128 + ch) * 256;
    v8f c = {};
    #pragma unroll
    for (int ks = 0; ks < 8; ++ks) {
      int kbase = ks * 32;
      int q = kbase >> 7;
      int cb = (kbase & 127) + 8 * hh;
      const _Float16* pa = sAct2 + (q * BM + ln) * 128;
      v16h a = combine8(pa + cb, pa + cb + 16);
      v16h b = combine8(pb + kbase + 8 * hh, pb + kbase + 16 + 8 * hh);
      c = wmma32(a, b, c);
    }
    float A = bnL[ch], C = bnL[128 + ch];
    #pragma unroll
    for (int r = 0; r < 8; ++r) {
      int ss = 8 * hh + r;
      sAct3[ss * 128 + ch] = (_Float16)mishf(A * c[r] + C);
    }
  }
  __syncthreads();

  // ---- stage 4: L3 GEMM  M=16 N=32 K=128 + clip/quant/accumulate
  if (wave < 2) {
    const float* bnL = bn + ((size_t)stack * 4 + 3) * 2 * 128;
    int ch = wave * 16 + ln;
    const _Float16* pb = wt3 + ((size_t)stack * 32 + ch) * 128;
    v8f c = {};
    #pragma unroll
    for (int ks = 0; ks < 4; ++ks) {
      int kbase = ks * 32;
      const _Float16* pa = sAct3 + ln * 128 + kbase + 8 * hh;
      v16h a = combine8(pa, pa + 16);
    v16h b = combine8(pb + kbase + 8 * hh, pb + kbase + 16 + 8 * hh);
      c = wmma32(a, b, c);
    }
    float A = bnL[ch], C = bnL[128 + ch];
    int fo = ori34 ? 0 : 32;
    #pragma unroll
    for (int r = 0; r < 8; ++r) {
      int ss = 8 * hh + r;
      float v = A * c[r] + C;
      v = fminf(fmaxf(v, -1.0f), QMAXF);
      v = rintf(v * 128.0f) * 0.0078125f;
      atomicAdd(&feat[(size_t)(s0 + ss) * 64 + fo + ch], v);
    }
  }
}

// ---------------------------------------------------------------------------
// MLP head: clip -> 64x32 -> clip -> 32x32 -> clip -> 32x3 (fake-quant W/b)
// ---------------------------------------------------------------------------
__global__ __launch_bounds__(256) void head_kernel(
    const float* __restrict__ feat,
    const float* __restrict__ w1, const float* __restrict__ b1,
    const float* __restrict__ w2, const float* __restrict__ b2,
    const float* __restrict__ w3, const float* __restrict__ b3,
    float* __restrict__ out)
{
  int s = blockIdx.x * blockDim.x + threadIdx.x;
  if (s >= NB) return;
  float v0[64], h1[32], h2[32];
  for (int i = 0; i < 64; ++i)
    v0[i] = fminf(fmaxf(feat[(size_t)s * 64 + i], -1.f), QMAXF);
  for (int o = 0; o < 32; ++o) {
    float acc = fq(b1[o]);
    for (int i = 0; i < 64; ++i) acc += fq(w1[o * 64 + i]) * v0[i];
    h1[o] = acc;
  }
  for (int i = 0; i < 32; ++i) h1[i] = fminf(fmaxf(h1[i], 0.f), QMAXF);
  for (int o = 0; o < 32; ++o) {
    float acc = fq(b2[o]);
    for (int i = 0; i < 32; ++i) acc += fq(w2[o * 32 + i]) * h1[i];
    h2[o] = acc;
  }
  for (int i = 0; i < 32; ++i) h2[i] = fminf(fmaxf(h2[i], 0.f), QMAXF);
  for (int j = 0; j < 3; ++j) {
    float acc = fq(b3[j]);
    for (int i = 0; i < 32; ++i) acc += fq(w3[j * 32 + i]) * h2[i];
    out[(size_t)s * 3 + j] = acc;
  }
}

// ---------------------------------------------------------------------------
// Host glue.  d_in indices follow jax pytree (sorted-key) flattening of the
// setup_inputs() dict:
//   0:b1 1:b2 2:b3
//   3..26  : p34 {b0..b3, be0..be3, g0..g3, m0..m3, v0..v3, w0..w3}
//   27..50 : p43 (same order)
//   51:w1 52:w2 53:w3 54:x
// ---------------------------------------------------------------------------
extern "C" void kernel_launch(void* const* d_in, const int* in_sizes, int n_in,
                              void* d_out, int out_size, void* d_ws, size_t ws_size,
                              hipStream_t stream)
{
  enum { I_B1 = 0, I_B2 = 1, I_B3 = 2, I_P34 = 3, I_P43 = 27,
         I_W1 = 51, I_W2 = 52, I_W3 = 53, I_X = 54 };

  auto getP = [&](int base) {
    StackParams P;
    for (int l = 0; l < 4; ++l) {
      P.b[l]  = (const float*)d_in[base + 0  + l];
      P.be[l] = (const float*)d_in[base + 4  + l];
      P.g[l]  = (const float*)d_in[base + 8  + l];
      P.m[l]  = (const float*)d_in[base + 12 + l];
      P.v[l]  = (const float*)d_in[base + 16 + l];
      P.w[l]  = (const float*)d_in[base + 20 + l];
    }
    return P;
  };
  StackParams P34 = getP(I_P34), P43 = getP(I_P43);

  char* ws = (char*)d_ws;
  _Float16* wt0 = (_Float16*)(ws + OFF_WT0);
  _Float16* wt1 = (_Float16*)(ws + OFF_WT1);
  _Float16* wt2 = (_Float16*)(ws + OFF_WT2);
  _Float16* wt3 = (_Float16*)(ws + OFF_WT3);
  float*    bnc = (float*)(ws + OFF_BN);
  float*    feat = (float*)(ws + OFF_FEAT);
  float*    out  = (float*)d_out;
  const float* x = (const float*)d_in[I_X];

  zero_kernel<<<1024, 256, 0, stream>>>(feat, out + (size_t)NB * 3,
                                        NB * 64, NB * 49);
  prep_kernel<<<dim3(4, NSTACK), 256, 0, stream>>>(P34, P43, wt0, wt1, wt2,
                                                   wt3, bnc);
  stack_kernel<<<dim3(NB / BM, NSTACK), 256, 0, stream>>>(x, wt0, wt1, wt2,
                                                          wt3, bnc, feat);
  head_kernel<<<NB / 256, 256, 0, stream>>>(
      feat, (const float*)d_in[I_W1], (const float*)d_in[I_B1],
      (const float*)d_in[I_W2], (const float*)d_in[I_B2],
      (const float*)d_in[I_W3], (const float*)d_in[I_B3], out);
}
